// S4DSSLayer_17420387352580
// MI455X (gfx1250) — matrix operand order repaired and checked by reading, not compiled, not run on hardware
//
#include <hip/hip_runtime.h>
#include <hip/hip_bf16.h>
#include <stdint.h>

// ---------------------------------------------------------------------------
// S4D layer as a chunked diagonal-SSM scan on CDNA5 WMMA (gfx1250, wave32).
//   out = circular_conv(x, K),  K[l] = Re( sum_n C_n r_n^l ),  r_n = exp(dt*Lambda_n)
// Per wave: 16 sequences, chunk T=32 timesteps:
//   out[32x16] = Ktoe[32x32] @ X[32x16]  +  Ccar[32x128] @ S[128x16]
//   S'[128x16] = Bmat[128x32] @ X[32x16] + rot(r^T) (x) S   (rot = WMMA C input)
// All GEMMs are v_wmma_f32_16x16x32_bf16 (K=32 exactly).
// State lives in VGPRs as 8 f32 16x16 D-tiles; the D->B fragment relayout for
// the carry GEMM is a half-wave swap: v_cvt_pk_bf16_f32 + __shfl_xor(16).
// L split into 8 segments with 2048-step warm-up (decay e^-10.2 ~ 3.6e-5);
// segment 0 warms up from the sequence tail => reproduces the circular wrap.
// ---------------------------------------------------------------------------

#define N_MODES   64
#define SEQLEN    16384
#define NSEQ      2048           // B*H = 8*256
#define T_CHUNK   32
#define SEGS      8
#define SEGLEN    (SEQLEN / SEGS)   // 2048
#define WARMUP    SEGLEN
#define GROUPS    (NSEQ / 16)       // 128 groups of 16 sequences
#define WPB       4                 // waves per block

// workspace byte offsets
#define WS_KTOE 0                   // 32*32 bf16  (2048 B)
#define WS_CCAR 2048                // 32*128 bf16 (8192 B)
#define WS_BMAT 10240               // 128*32 bf16 (8192 B)
#define WS_AB   18432               // 64 float2   (512 B)  r_n^T

typedef __attribute__((ext_vector_type(16))) __bf16 v16bf;
typedef __attribute__((ext_vector_type(8)))  float  v8f;
typedef uint32_t u32;

static __device__ inline v8f vz8() {
  v8f z;
#pragma unroll
  for (int i = 0; i < 8; ++i) z[i] = 0.f;
  return z;
}

// pack two f32 into one dword of bf16 pairs (lowers to v_cvt_pk_bf16_f32)
static __device__ inline u32 pk2bf(float a, float b) {
  union { __bf16 h[2]; u32 u; } t;
  t.h[0] = (__bf16)a; t.h[1] = (__bf16)b;
  return t.u;
}

// ---------------------------------------------------------------------------
// Setup kernel: build Ktoe / Ccar / Bmat / r^T tables from Lambda, w, dt.
// ---------------------------------------------------------------------------
__global__ void s4d_build_tables(const float* __restrict__ Lr,
                                 const float* __restrict__ Li,
                                 const float* __restrict__ wr,
                                 const float* __restrict__ wi,
                                 const float* __restrict__ dtp,
                                 char* __restrict__ ws) {
  __shared__ float Ks[T_CHUNK];
  const int n = threadIdx.x;
  if (n < T_CHUNK) Ks[n] = 0.f;
  __syncthreads();

  __bf16* ktoe = (__bf16*)(ws + WS_KTOE);
  __bf16* ccar = (__bf16*)(ws + WS_CCAR);
  __bf16* bmat = (__bf16*)(ws + WS_BMAT);
  float2* ab   = (float2*)(ws + WS_AB);

  if (n < N_MODES) {
    const float dt = dtp[0];
    const float lr = Lr[n], li = Li[n];
    const float zr = lr * dt, zi = li * dt;
    const float er = __expf(zr);
    const float rre = er * __cosf(zi), rim = er * __sinf(zi);   // r = e^z
    // C = w * (e^z - 1) / Lambda
    const float e1r = rre - 1.f, e1i = rim;
    const float numr = wr[n] * e1r - wi[n] * e1i;
    const float numi = wr[n] * e1i + wi[n] * e1r;
    const float den  = lr * lr + li * li;
    const float Cr = (numr * lr + numi * li) / den;
    const float Ci = (numi * lr - numr * li) / den;

    // K[d] = Re(C r^d), reduced over modes via ds_add_f32
    float pr = Cr, pi = Ci;
    for (int d = 0; d < T_CHUNK; ++d) {
      atomicAdd(&Ks[d], pr);
      const float t = pr * rre - pi * rim; pi = pr * rim + pi * rre; pr = t;
    }
    // Ccar[t][2n]=Re(C r^{t+1}), [t][2n+1]=-Im(C r^{t+1})
    pr = Cr * rre - Ci * rim; pi = Cr * rim + Ci * rre;
    for (int t = 0; t < T_CHUNK; ++t) {
      ccar[t * 128 + 2 * n]     = (__bf16)pr;
      ccar[t * 128 + 2 * n + 1] = (__bf16)(-pi);
      const float tm = pr * rre - pi * rim; pi = pr * rim + pi * rre; pr = tm;
    }
    // Bmat[2n][t]=Re(r^{T-1-t}), Bmat[2n+1][t]=Im(r^{T-1-t});  q ends at r^T
    float qr = 1.f, qi = 0.f;
    for (int tt = 0; tt < T_CHUNK; ++tt) {
      const int t = T_CHUNK - 1 - tt;
      bmat[(2 * n)     * T_CHUNK + t] = (__bf16)qr;
      bmat[(2 * n + 1) * T_CHUNK + t] = (__bf16)qi;
      const float tm = qr * rre - qi * rim; qi = qr * rim + qi * rre; qr = tm;
    }
    ab[n] = make_float2(qr, qi);   // r^T
  }
  __syncthreads();
  if (n < T_CHUNK) {
    const int i = n;
    for (int j = 0; j < T_CHUNK; ++j)
      ktoe[i * T_CHUNK + j] = (j <= i) ? (__bf16)Ks[i - j] : (__bf16)0.f;
  }
}

// ---------------------------------------------------------------------------
// Fragment helpers (wave32; A/B in this code's contiguous k order, C/D per ISA 7.12.2).
// A 16x32 bf16: lane<16 -> row=lane, K = j; lane>=16: K+=16.
// B 32x16 bf16: element j -> K = j + 16*hi, N = lane&15.
// C/D 16x16 f32: element v -> M = v + 8*hi,  N = lane&15.
// ---------------------------------------------------------------------------
static __device__ inline v16bf load_afrag(const __bf16* __restrict__ mat,
                                          int row0, int ldm, int kbase, int lane) {
  v16bf f;
  const int row  = row0 + (lane & 15);
  const int koff = (lane >= 16) ? 16 : 0;
#pragma unroll
  for (int j = 0; j < 16; ++j) {
    const int k = kbase + koff + j;
    f[j] = mat[row * ldm + k];
  }
  return f;
}

static __device__ inline v16bf load_xfrag(const float* __restrict__ xs,
                                          int t0, int hi) {
  // B-fragment of X[32 time x 16 seq]: lane's column is its sequence,
  // 16 contiguous timesteps per lane -> 4x global_load_b128, fully coalesced.
  const float4* p = (const float4*)(xs + t0 + hi * 16);
  v16bf f;
#pragma unroll
  for (int q = 0; q < 4; ++q) {
    const float4 v = p[q];
    f[4 * q + 0] = (__bf16)v.x;
    f[4 * q + 1] = (__bf16)v.y;
    f[4 * q + 2] = (__bf16)v.z;
    f[4 * q + 3] = (__bf16)v.w;
  }
  return f;
}

// Build carry B-fragment (state rows 32*kt .. 32*kt+31) from two f32 D-tiles.
// D tile mt=2kt holds rows 32kt + v + 8*hi; tile mt=2kt+1 holds +16.
// B frag needs rows 32kt + 16*hi + j  => half-wave swap with lane^16.
static __device__ inline v16bf make_sfrag(const v8f& tA, const v8f& tB, int hi) {
  u32 pA[4], pB[4], recv[4];
#pragma unroll
  for (int q = 0; q < 4; ++q) {
    pA[q] = pk2bf(tA[2 * q], tA[2 * q + 1]);
    pB[q] = pk2bf(tB[2 * q], tB[2 * q + 1]);
  }
#pragma unroll
  for (int q = 0; q < 4; ++q) {
    const u32 send = hi ? pA[q] : pB[q];   // give partner what it needs
    recv[q] = (u32)__shfl_xor((int)send, 16, 32);
  }
  union { u32 d[8]; v16bf f; } u;
#pragma unroll
  for (int q = 0; q < 4; ++q) {
    u.d[q]     = hi ? recv[q] : pA[q];     // rows 32kt+16hi+0..7
    u.d[4 + q] = hi ? pB[q]   : recv[q];   // rows 32kt+16hi+8..15
  }
  return u.f;
}

static __device__ inline void store_out(float* __restrict__ os, int t0,
                                        v8f acc, int m, int hi) {
  const int t = t0 + 16 * m + hi * 8;      // D layout: element v -> time t+v
  *(float4*)(os + t)     = make_float4(acc[0], acc[1], acc[2], acc[3]);
  *(float4*)(os + t + 4) = make_float4(acc[4], acc[5], acc[6], acc[7]);
}

// S' = Bmat @ X + rot(r^T)(x)S : rotation feeds the WMMA C operand directly.
static __device__ inline void step_state(v16bf xf, v8f* __restrict__ st,
                                         const float2* __restrict__ ab,
                                         const v16bf* __restrict__ bmatF,
                                         int hi) {
#pragma unroll
  for (int mt = 0; mt < 8; ++mt) {
    v8f rot;
#pragma unroll
    for (int p = 0; p < 4; ++p) {
      const int n = 8 * mt + hi * 4 + p;   // mode; uniform per half-wave
      const float2 rt = ab[n];             // LDS broadcast read
      const float sr = st[mt][2 * p], si = st[mt][2 * p + 1];
      rot[2 * p]     = rt.x * sr - rt.y * si;
      rot[2 * p + 1] = rt.y * sr + rt.x * si;
    }
    st[mt] = __builtin_amdgcn_wmma_f32_16x16x32_bf16(
        false, bmatF[mt], false, xf, (short)0, rot, false, false);
  }
}

// ---------------------------------------------------------------------------
// Main scan kernel: 1024 waves = 128 seq-groups x 8 segments.
// ---------------------------------------------------------------------------
__global__ void __launch_bounds__(WPB * 32)
s4d_scan(const float* __restrict__ x, float* __restrict__ out,
         const char* __restrict__ ws) {
  __shared__ float2 ldsAB[N_MODES];               // r^T per mode (512 B)

  const int tid = threadIdx.x;
  if (tid < N_MODES) ldsAB[tid] = ((const float2*)(ws + WS_AB))[tid];

  const int lane  = tid & 31;
  const int wslot = tid >> 5;
  const int wgid  = blockIdx.x * WPB + wslot;     // 0..1023
  const int seg   = wgid & (SEGS - 1);
  const int grp   = wgid / SEGS;                  // 0..127
  const int col   = lane & 15;
  const int hi    = (lane >= 16) ? 1 : 0;

  const long seq = (long)grp * 16 + col;
  const float* xs = x   + seq * (long)SEQLEN;
  float*       os = out + seq * (long)SEQLEN;
  __syncthreads();

  // Constant A-fragments resident in VGPRs for the whole scan.
  const __bf16* ktoe = (const __bf16*)(ws + WS_KTOE);
  const __bf16* ccar = (const __bf16*)(ws + WS_CCAR);
  const __bf16* bmat = (const __bf16*)(ws + WS_BMAT);
  v16bf ktoeF[2], ccarF[2][4], bmatF[8];
#pragma unroll
  for (int m = 0; m < 2; ++m) ktoeF[m] = load_afrag(ktoe, 16 * m, 32, 0, lane);
#pragma unroll
  for (int m = 0; m < 2; ++m)
#pragma unroll
    for (int kt = 0; kt < 4; ++kt)
      ccarF[m][kt] = load_afrag(ccar, 16 * m, 128, 32 * kt, lane);
#pragma unroll
  for (int mt = 0; mt < 8; ++mt) bmatF[mt] = load_afrag(bmat, 16 * mt, 32, 0, lane);

  // State: 8 f32 16x16 D-tiles in VGPRs (rows 16mt+v+8hi, col = lane&15).
  v8f st[8];
#pragma unroll
  for (int mt = 0; mt < 8; ++mt) st[mt] = vz8();

  const int segstart = seg * SEGLEN;

  // Warm-up over [segstart-WARMUP, segstart) mod L: state only.
  // Seg 0 therefore reads the tail of the sequence -> circular convolution.
  for (int c = 0; c < WARMUP / T_CHUNK; ++c) {
    int t0 = segstart - WARMUP + c * T_CHUNK;
    if (t0 < 0) t0 += SEQLEN;
    const v16bf xf = load_xfrag(xs, t0, hi);
    step_state(xf, st, ldsAB, bmatF, hi);
  }

  // Main segment: 18 WMMA per 32 timesteps x 16 sequences.
  for (int c = 0; c < SEGLEN / T_CHUNK; ++c) {
    const int t0 = segstart + c * T_CHUNK;
    if (t0 + 5 * T_CHUNK <= SEQLEN)
      __builtin_prefetch(xs + t0 + 4 * T_CHUNK + hi * 16, 0, 0);

    const v16bf xf = load_xfrag(xs, t0, hi);

    v8f acc0 = vz8(), acc1 = vz8();
#pragma unroll
    for (int kt = 0; kt < 4; ++kt) {               // carry: Ccar @ S
      const v16bf sf = make_sfrag(st[2 * kt], st[2 * kt + 1], hi);
      acc0 = __builtin_amdgcn_wmma_f32_16x16x32_bf16(
          false, ccarF[0][kt], false, sf, (short)0, acc0, false, false);
      acc1 = __builtin_amdgcn_wmma_f32_16x16x32_bf16(
          false, ccarF[1][kt], false, sf, (short)0, acc1, false, false);
    }
    // intra-chunk Toeplitz: Ktoe @ X
    acc0 = __builtin_amdgcn_wmma_f32_16x16x32_bf16(
        false, ktoeF[0], false, xf, (short)0, acc0, false, false);
    acc1 = __builtin_amdgcn_wmma_f32_16x16x32_bf16(
        false, ktoeF[1], false, xf, (short)0, acc1, false, false);

    store_out(os, t0, acc0, 0, hi);
    store_out(os, t0, acc1, 1, hi);

    step_state(xf, st, ldsAB, bmatF, hi);
  }
}

// ---------------------------------------------------------------------------
extern "C" void kernel_launch(void* const* d_in, const int* in_sizes, int n_in,
                              void* d_out, int out_size, void* d_ws, size_t ws_size,
                              hipStream_t stream) {
  (void)in_sizes; (void)n_in; (void)out_size; (void)ws_size;
  const float* x  = (const float*)d_in[0];
  const float* Lr = (const float*)d_in[1];
  const float* Li = (const float*)d_in[2];
  const float* wr = (const float*)d_in[3];
  const float* wi = (const float*)d_in[4];
  const float* dt = (const float*)d_in[5];

  s4d_build_tables<<<1, 64, 0, stream>>>(Lr, Li, wr, wi, dt, (char*)d_ws);
  s4d_scan<<<GROUPS * SEGS / WPB, WPB * 32, 0, stream>>>(
      x, (float*)d_out, (const char*)d_ws);
}
